// CyclicNoiseOscillator_87471303950763
// MI455X (gfx1250) — matrix operand order repaired and checked by reading, not compiled, not run on hardware
//
#include <hip/hip_runtime.h>

// ---------------------------------------------------------------------------
// CyclicNoiseOscillator for MI455X (gfx1250, wave32).
//
// Conv-as-WMMA: out[t0+16i+n] = sum_a sum_kk A_a[i][kk] * B_a[kk][n]
//   A_a[i][kk] = impulse[t0 + 16a + 16i + kk]        (16x32 f16 tile)
//   B_a[kk][n] = h[16a + kk - n] if 0<=kk-n<16 else 0 (32x16 banded tile)
// 3-pass hi/lo f16 split recovers ~f32 accuracy.
// k_conv: each wave computes TWO 256-output tiles in one a-loop with
// 3 accumulators per tile -> 6 independent v_wmma chains (no hazard stalls)
// and B fragments loaded once per a-step for both tiles.
//
// Workspace layout (bytes):
//   [0      ..  8192) blockSums  16*125 f32
//   [8192   .. 32768) blockOff   16*126 f32
//   [32768  ..      ) xh   8,192,000 f16  (16,384,000 B)
//   [+16384000      ) xl   8,192,000 f16
//   [+2*16384000    ) Bhi  8192 f16 (16 KB), then Blo 8192 f16
// Total ~32.83 MB of d_ws.
// ---------------------------------------------------------------------------

typedef _Float16 h8  __attribute__((ext_vector_type(8)));
typedef _Float16 h16 __attribute__((ext_vector_type(16)));
typedef float    f8  __attribute__((ext_vector_type(8)));

#define SRATE    24000.0f
#define KS       250
#define NROW     16
#define LFR      2000
#define TROW     512000
#define SCAN_BLK 4096
#define NBLK     125        // TROW / SCAN_BLK
#define TPB      256

// Linear-upsample increment f0_up(t)/SR, matching jax _linear_upsample
// (align_corners=False, clamped indices).
__device__ __forceinline__ float f0_inc(const float* __restrict__ f0row, int t) {
    float src = ((float)t - 127.5f) * (1.0f / 256.0f);
    src = fmaxf(src, 0.0f);
    int i0 = (int)src;                       // src >= 0 -> trunc == floor
    if (i0 > LFR - 1) i0 = LFR - 1;
    int i1 = i0 + 1; if (i1 > LFR - 1) i1 = LFR - 1;
    float w = src - (float)i0;
    float v = f0row[i0] * (1.0f - w) + f0row[i1] * w;
    return v * (1.0f / SRATE);
}

// --- K1: per-(row, 4096-sample block) sum of f0_up/SR --------------------
__global__ void __launch_bounds__(TPB) k_blocksum(const float* __restrict__ f0,
                                                  float* __restrict__ blockSums) {
    int row = blockIdx.y, b = blockIdx.x, tid = threadIdx.x;
    const float* f0row = f0 + row * LFR;
    int t0 = b * SCAN_BLK + tid * 16;
    float s = 0.0f;
#pragma unroll
    for (int j = 0; j < 16; ++j) s += f0_inc(f0row, t0 + j);
    __shared__ float red[TPB];
    red[tid] = s; __syncthreads();
    for (int off = TPB / 2; off > 0; off >>= 1) {
        if (tid < off) red[tid] += red[tid + off];
        __syncthreads();
    }
    if (tid == 0) blockSums[row * NBLK + b] = red[0];
}

// --- K2: per-row exclusive scan of the 125 block sums (+ row total) ------
__global__ void k_scanblocks(const float* __restrict__ blockSums,
                             float* __restrict__ blockOff) {
    int r = threadIdx.x;
    if (r < NROW) {
        float acc = 0.0f;
        for (int j = 0; j < NBLK; ++j) {
            blockOff[r * (NBLK + 1) + j] = acc;
            acc += blockSums[r * NBLK + j];
        }
        blockOff[r * (NBLK + 1) + NBLK] = acc;   // row total (for jnp.roll wrap)
    }
}

// --- K3: rad = -cumsum, sawtooth, impulse; emit f16 hi/lo ----------------
__global__ void __launch_bounds__(TPB) k_impulse(const float* __restrict__ f0,
                                                 const float* __restrict__ blockOff,
                                                 _Float16* __restrict__ xh,
                                                 _Float16* __restrict__ xl) {
    int row = blockIdx.y, b = blockIdx.x, tid = threadIdx.x;
    const float* f0row = f0 + row * LFR;
    int t0 = b * SCAN_BLK + tid * 16;

    float inc[16];
    float s = 0.0f;
#pragma unroll
    for (int j = 0; j < 16; ++j) { inc[j] = f0_inc(f0row, t0 + j); s += inc[j]; }

    __shared__ float sc[TPB];
    sc[tid] = s; __syncthreads();
    for (int off = 1; off < TPB; off <<= 1) {        // Hillis-Steele inclusive
        float v = (tid >= off) ? sc[tid - off] : 0.0f;
        __syncthreads();
        sc[tid] += v;
        __syncthreads();
    }
    float exclT = sc[tid] - s;

    float base = blockOff[row * (NBLK + 1) + b] + exclT;   // positive cumsum before me
    float prevRad = -base;
    if (b == 0 && tid == 0)                                 // jnp.roll: t=0 sees t=T-1
        prevRad = -blockOff[row * (NBLK + 1) + NBLK];
    float prevSaw = prevRad - floorf(prevRad);

    h8 vh0, vh1, vl0, vl1;
    float run = 0.0f;
#pragma unroll
    for (int j = 0; j < 16; ++j) {
        run += inc[j];
        float rad = -(base + run);
        float saw = rad - floorf(rad);                      // jnp.mod(rad, 1.0)
        float imp = saw - prevSaw;
        prevSaw = saw;
        _Float16 hi = (_Float16)imp;
        _Float16 lo = (_Float16)(imp - (float)hi);
        if (j < 8) { vh0[j] = hi; vl0[j] = lo; }
        else       { vh1[j - 8] = hi; vl1[j - 8] = lo; }
    }
    size_t o = (size_t)row * TROW + t0;
    *(h8*)(xh + o)     = vh0;
    *(h8*)(xh + o + 8) = vh1;
    *(h8*)(xl + o)     = vl0;
    *(h8*)(xl + o + 8) = vl1;
}

// --- K4: pack banded B fragments in wave32 WMMA B layout -----------------
// B element (kk, n): lane = n + 16*((kk>>3)&1), half-slot = (kk&7) + 8*(kk>>4).
// Stored as Bws[(a*32 + lane)*16 + slot].
__global__ void k_pack(const float* __restrict__ kern,
                       _Float16* __restrict__ Bhi, _Float16* __restrict__ Blo) {
    for (int i = threadIdx.x; i < 16 * 32 * 16; i += blockDim.x) {
        int a    = i >> 9;
        int lane = (i >> 4) & 31;
        int slot = i & 15;
        int kk   = ((slot >> 3) << 4) + ((lane >> 4) << 3) + (slot & 7);
        int n    = lane & 15;
        int c    = kk - n;
        float v  = 0.0f;
        int  k   = 16 * a + c;
        if (c >= 0 && c < 16 && k < KS) v = kern[k];
        _Float16 hi = (_Float16)v;
        Bhi[i] = hi;
        Blo[i] = (_Float16)(v - (float)hi);
    }
}

// --- K5: WMMA conv + voiced/noise mix ------------------------------------
__global__ void __launch_bounds__(TPB) k_conv(const _Float16* __restrict__ xh,
                                              const _Float16* __restrict__ xl,
                                              const _Float16* __restrict__ Bhi,
                                              const _Float16* __restrict__ Blo,
                                              const float* __restrict__ uv,
                                              const float* __restrict__ noise,
                                              float* __restrict__ out) {
    __shared__ __align__(16) _Float16 sxh[4608];   // 4096 outputs + 511 halo
    __shared__ __align__(16) _Float16 sxl[4608];

    int row = blockIdx.y, b = blockIdx.x, tid = threadIdx.x;
    size_t rowOff = (size_t)row * TROW;
    int t0b = b * SCAN_BLK;
    const _Float16* gxh = xh + rowOff;
    const _Float16* gxl = xl + rowOff;

    for (int i = tid; i < 4608 / 8; i += TPB) {
        int gi = t0b + i * 8;
        h8 vh, vl;
        if (gi + 8 <= TROW) {
            vh = *(const h8*)(gxh + gi);
            vl = *(const h8*)(gxl + gi);
        } else {
#pragma unroll
            for (int j = 0; j < 8; ++j) {
                vh[j] = (gi + j < TROW) ? gxh[gi + j] : (_Float16)0.0f;
                vl[j] = (gi + j < TROW) ? gxl[gi + j] : (_Float16)0.0f;
            }
        }
        *(h8*)(sxh + i * 8) = vh;
        *(h8*)(sxl + i * 8) = vl;
    }
    __syncthreads();

    int lane = tid & 31;
    int wave = tid >> 5;
    int m    = lane & 15;
    int g    = lane >> 4;
    int off8 = g * 8;          // A layout: lane group 1 starts at K=8 within each 16-K span

    // Two 256-output tiles per wave, computed in a single a-loop.
    int t0locA = wave * 512;               // tile A local base; tile B = +256
    int abase  = t0locA + 16 * m + off8;

    h8 qh0A = *(const h8*)(sxh + abase);
    h8 ql0A = *(const h8*)(sxl + abase);
    h8 qh0B = *(const h8*)(sxh + abase + 256);
    h8 ql0B = *(const h8*)(sxl + abase + 256);

    const f8 z = {0.0f, 0.0f, 0.0f, 0.0f, 0.0f, 0.0f, 0.0f, 0.0f};
    f8 aHHA = z, aHLA = z, aLHA = z;       // 3 independent chains, tile A
    f8 aHHB = z, aHLB = z, aLHB = z;       // 3 independent chains, tile B

#pragma unroll
    for (int a = 0; a < 16; ++a) {
        int s = abase + 16 * a + 16;
        h8 qh1A = *(const h8*)(sxh + s);
        h8 ql1A = *(const h8*)(sxl + s);
        h8 qh1B = *(const h8*)(sxh + s + 256);
        h8 ql1B = *(const h8*)(sxl + s + 256);

        h16 AhA = __builtin_shufflevector(qh0A, qh1A, 0,1,2,3,4,5,6,7,8,9,10,11,12,13,14,15);
        h16 AlA = __builtin_shufflevector(ql0A, ql1A, 0,1,2,3,4,5,6,7,8,9,10,11,12,13,14,15);
        h16 AhB = __builtin_shufflevector(qh0B, qh1B, 0,1,2,3,4,5,6,7,8,9,10,11,12,13,14,15);
        h16 AlB = __builtin_shufflevector(ql0B, ql1B, 0,1,2,3,4,5,6,7,8,9,10,11,12,13,14,15);

        // Tile-invariant B fragments: loaded once, used by both tiles.
        const _Float16* bph = Bhi + (a * 32 + lane) * 16;
        const _Float16* bpl = Blo + (a * 32 + lane) * 16;
        h8 bh0 = *(const h8*)bph;       h8 bh1 = *(const h8*)(bph + 8);
        h8 bl0 = *(const h8*)bpl;       h8 bl1 = *(const h8*)(bpl + 8);
        h16 Bh = __builtin_shufflevector(bh0, bh1, 0,1,2,3,4,5,6,7,8,9,10,11,12,13,14,15);
        h16 Bl = __builtin_shufflevector(bl0, bl1, 0,1,2,3,4,5,6,7,8,9,10,11,12,13,14,15);

        aHHA = __builtin_amdgcn_wmma_f32_16x16x32_f16(false, AhA, false, Bh, (short)0, aHHA, false, false);
        aHHB = __builtin_amdgcn_wmma_f32_16x16x32_f16(false, AhB, false, Bh, (short)0, aHHB, false, false);
        aHLA = __builtin_amdgcn_wmma_f32_16x16x32_f16(false, AhA, false, Bl, (short)0, aHLA, false, false);
        aHLB = __builtin_amdgcn_wmma_f32_16x16x32_f16(false, AhB, false, Bl, (short)0, aHLB, false, false);
        aLHA = __builtin_amdgcn_wmma_f32_16x16x32_f16(false, AlA, false, Bh, (short)0, aLHA, false, false);
        aLHB = __builtin_amdgcn_wmma_f32_16x16x32_f16(false, AlB, false, Bh, (short)0, aLHB, false, false);

        qh0A = qh1A; ql0A = ql1A;
        qh0B = qh1B; ql0B = ql1B;
    }

    f8 accA = aHHA + aHLA + aLHA;
    f8 accB = aHHB + aHLB + aLHB;

    // D layout: VGPR r, lanes 0-15 -> (M=r, N=lane); lanes 16-31 -> (M=r+8, N=lane-16)
    int n = lane & 15;
    const float* uvrow = uv + row * LFR;

    int t0A = t0b + t0locA;
#pragma unroll
    for (int r = 0; r < 8; ++r) {
        int t = t0A + 16 * r + 128 * g + n;
        float uvv = uvrow[t >> 8];                         // nearest (repeat) upsample
        out[rowOff + t] = accA[r] * uvv + (1.0f - uvv) * noise[rowOff + t];
    }
    int t0B = t0A + 256;
#pragma unroll
    for (int r = 0; r < 8; ++r) {
        int t = t0B + 16 * r + 128 * g + n;
        float uvv = uvrow[t >> 8];
        out[rowOff + t] = accB[r] * uvv + (1.0f - uvv) * noise[rowOff + t];
    }
}

// ---------------------------------------------------------------------------
extern "C" void kernel_launch(void* const* d_in, const int* in_sizes, int n_in,
                              void* d_out, int out_size, void* d_ws, size_t ws_size,
                              hipStream_t stream) {
    (void)in_sizes; (void)n_in; (void)out_size; (void)ws_size;
    const float* f0    = (const float*)d_in[0];
    const float* uvp   = (const float*)d_in[1];
    const float* noise = (const float*)d_in[2];
    const float* kern  = (const float*)d_in[3];
    float* out = (float*)d_out;

    char* ws = (char*)d_ws;
    float*    blockSums = (float*)(ws);
    float*    blockOff  = (float*)(ws + 8192);
    _Float16* xh        = (_Float16*)(ws + 32768);
    _Float16* xl        = (_Float16*)(ws + 32768 + 16384000);
    _Float16* Bhi       = (_Float16*)(ws + 32768 + 2 * 16384000);
    _Float16* Blo       = (_Float16*)(ws + 32768 + 2 * 16384000 + 16384);

    dim3 grid(NBLK, NROW);
    k_blocksum <<<grid, TPB, 0, stream>>>(f0, blockSums);
    k_scanblocks<<<1, 32, 0, stream>>>(blockSums, blockOff);
    k_impulse  <<<grid, TPB, 0, stream>>>(f0, blockOff, xh, xl);
    k_pack     <<<1, TPB, 0, stream>>>(kern, Bhi, Blo);
    k_conv     <<<grid, TPB, 0, stream>>>(xh, xl, Bhi, Blo, uvp, noise, out);
}
